// ComplexGRUCell_17652315587081
// MI455X (gfx1250) — matrix operand order repaired and checked by reading, not compiled, not run on hardware
//
#include <hip/hip_runtime.h>
#include <cstdint>
#include <cstddef>

typedef __bf16 bf16_t;
typedef __attribute__((ext_vector_type(16))) __bf16 v16bf;
typedef __attribute__((ext_vector_type(8)))  __bf16 v8bf;
typedef __attribute__((ext_vector_type(8)))  float   v8f;

#define TILE_M 128
#define TILE_N 64
#define TILE_K 32
#define LDSPAD 40   // bf16 elements per LDS row (80B, 16B-aligned, dodges bank conflicts)

static __device__ __forceinline__ v16bf frag_from_halves(v8bf lo, v8bf hi) {
  v16bf r;
#pragma unroll
  for (int i = 0; i < 8; ++i) { r[i] = lo[i]; r[i + 8] = hi[i]; }
  return r;
}

static __device__ __forceinline__ float sigmoidf_dev(float x) {
  return 1.0f / (1.0f + __expf(-x));
}

// CDNA5 async copy: 16B global -> LDS, tracked by ASYNCcnt, no VGPR data path.
// LDS operand is the wave-relative LDS byte offset == low 32 bits of the
// generic pointer.
static __device__ __forceinline__ void async_copy_b128(void* lds_ptr, const void* gptr) {
  const uint32_t lds_addr = (uint32_t)(uintptr_t)lds_ptr;
  const uint64_t gaddr    = (uint64_t)(uintptr_t)gptr;
  asm volatile("global_load_async_to_lds_b128 %0, %1, off"
               :: "v"(lds_addr), "v"(gaddr) : "memory");
}

static __device__ __forceinline__ void wait_async_le6() {
  asm volatile("s_wait_asynccnt 0x6" ::: "memory");
}
static __device__ __forceinline__ void wait_async_0() {
  asm volatile("s_wait_asynccnt 0x0" ::: "memory");
}

// ---------------------------------------------------------------------------
// Complex GEMM:  C[m,n] = sum_k A[m,k] * W[n,k]   (complex, bf16 in / f32 out)
// Two K-segments (A0/W0:K0 then A1/W1:K1) accumulate into the same tile, plus
// up to two complex bias rows and an elementwise add-in.
// Block = 256 threads = 8 waves; block tile 128x64; wave tile 32x32.
// Double-buffered LDS fed by GLOBAL_LOAD_ASYNC_TO_LDS_B128.
// ---------------------------------------------------------------------------
__global__ __launch_bounds__(256)
void cgru_gemm_cplx(const bf16_t* __restrict__ A0re, const bf16_t* __restrict__ A0im,
                    const bf16_t* __restrict__ W0re, const bf16_t* __restrict__ W0im, int K0,
                    const bf16_t* __restrict__ A1re, const bf16_t* __restrict__ A1im,
                    const bf16_t* __restrict__ W1re, const bf16_t* __restrict__ W1im, int K1,
                    const float* __restrict__ b1re, const float* __restrict__ b1im,
                    const float* __restrict__ b2re, const float* __restrict__ b2im,
                    const float* __restrict__ addre, const float* __restrict__ addim,
                    float* __restrict__ Cre, float* __restrict__ Cim, int N)
{
  // [stage][re/im][row][k]
  __shared__ __align__(16) bf16_t As[2][2][TILE_M][LDSPAD];
  __shared__ __align__(16) bf16_t Bs[2][2][TILE_N][LDSPAD];

  const int tid  = threadIdx.x;
  const int lane = tid & 31;
  const int wv   = tid >> 5;
  const int wm   = wv >> 1;          // wave M index 0..3
  const int wn   = wv & 1;           // wave N index 0..1
  const int m0   = blockIdx.x * TILE_M;
  const int n0   = blockIdx.y * TILE_N;
  const int lh   = lane >> 4;        // lane half (0/1)
  const int ll   = lane & 15;

  const v8f vzero = {0.f, 0.f, 0.f, 0.f, 0.f, 0.f, 0.f, 0.f};
  v8f acc_re[2][2], acc_im[2][2];
#pragma unroll
  for (int i = 0; i < 2; ++i)
#pragma unroll
    for (int j = 0; j < 2; ++j) { acc_re[i][j] = vzero; acc_im[i][j] = vzero; }

  const bf16_t* Apl[2][2] = { {A0re, A0im}, {A1re, A1im} };
  const bf16_t* Wpl[2][2] = { {W0re, W0im}, {W1re, W1im} };
  const int Ks[2] = { K0, K1 };
  const int nt0 = K0 >> 5;
  const int nt  = nt0 + (K1 >> 5);

  const int ldr = tid >> 2;          // 0..63 : staging row
  const int ldc = (tid & 3) * 8;     // 0,8,16,24 : staging col (bf16 elems)

  // Issue the 6 async 16B transfers for logical tile t into LDS stage s.
  auto issue = [&](int s, int t) {
    const int seg = (t < nt0) ? 0 : 1;
    const int k0  = ((seg == 0) ? t : (t - nt0)) << 5;
    const int K   = Ks[seg];
    const bf16_t* Ar = Apl[seg][0];
    const bf16_t* Ai = Apl[seg][1];
    const bf16_t* Wr = Wpl[seg][0];
    const bf16_t* Wi = Wpl[seg][1];
    const size_t arow0 = (size_t)(m0 + ldr)      * (size_t)K + (size_t)(k0 + ldc);
    const size_t arow1 = (size_t)(m0 + 64 + ldr) * (size_t)K + (size_t)(k0 + ldc);
    const size_t wrow  = (size_t)(n0 + ldr)      * (size_t)K + (size_t)(k0 + ldc);
    async_copy_b128(&As[s][0][ldr     ][ldc], &Ar[arow0]);
    async_copy_b128(&As[s][0][ldr + 64][ldc], &Ar[arow1]);
    async_copy_b128(&As[s][1][ldr     ][ldc], &Ai[arow0]);
    async_copy_b128(&As[s][1][ldr + 64][ldc], &Ai[arow1]);
    async_copy_b128(&Bs[s][0][ldr][ldc], &Wr[wrow]);
    async_copy_b128(&Bs[s][1][ldr][ldc], &Wi[wrow]);
  };

  int buf = 0;
  issue(0, 0);

  for (int t = 0; t < nt; ++t) {
    if (t + 1 < nt) {
      issue(buf ^ 1, t + 1);   // prefetch next tile into the other stage
      wait_async_le6();        // first batch (stage `buf`) has fully landed
    } else {
      wait_async_0();
    }
    __syncthreads();

    // Fragments per the CDNA5 WMMA 16-bit VGPR layouts (ISA 7.12.2).
    v16bf ar[2], ai[2], nai[2], br[2], bi[2];
#pragma unroll
    for (int mi = 0; mi < 2; ++mi) {
      const bf16_t* rowr = &As[buf][0][wm * 32 + mi * 16 + ll][0];
      const bf16_t* rowi = &As[buf][1][wm * 32 + mi * 16 + ll][0];
      // A 16x32: lane half 0 -> K 0-7 & 16-23 ; half 1 -> K 8-15 & 24-31
      ar[mi] = frag_from_halves(*(const v8bf*)&rowr[lh * 8],
                                *(const v8bf*)&rowr[16 + lh * 8]);
      ai[mi] = frag_from_halves(*(const v8bf*)&rowi[lh * 8],
                                *(const v8bf*)&rowi[16 + lh * 8]);
#pragma unroll
      for (int e = 0; e < 16; ++e) nai[mi][e] = (bf16_t)(-(float)ai[mi][e]);
    }
#pragma unroll
    for (int ni = 0; ni < 2; ++ni) {
      const bf16_t* rowr = &Bs[buf][0][wn * 32 + ni * 16 + ll][0];
      const bf16_t* rowi = &Bs[buf][1][wn * 32 + ni * 16 + ll][0];
      // B 32x16: lane half 0 -> K 0-7 & 16-23 ; half 1 -> K 8-15 & 24-31 (col = lane%16)
      br[ni] = frag_from_halves(*(const v8bf*)&rowr[lh * 8],
                                *(const v8bf*)&rowr[16 + lh * 8]);
      bi[ni] = frag_from_halves(*(const v8bf*)&rowi[lh * 8],
                                *(const v8bf*)&rowi[16 + lh * 8]);
    }

#pragma unroll
    for (int mi = 0; mi < 2; ++mi)
#pragma unroll
      for (int ni = 0; ni < 2; ++ni) {
        // C_re += Ar*Br - Ai*Bi ; C_im += Ar*Bi + Ai*Br
        acc_re[mi][ni] = __builtin_amdgcn_wmma_f32_16x16x32_bf16(
            false, ar[mi],  false, br[ni], (short)0, acc_re[mi][ni], false, false);
        acc_re[mi][ni] = __builtin_amdgcn_wmma_f32_16x16x32_bf16(
            false, nai[mi], false, bi[ni], (short)0, acc_re[mi][ni], false, false);
        acc_im[mi][ni] = __builtin_amdgcn_wmma_f32_16x16x32_bf16(
            false, ar[mi],  false, bi[ni], (short)0, acc_im[mi][ni], false, false);
        acc_im[mi][ni] = __builtin_amdgcn_wmma_f32_16x16x32_bf16(
            false, ai[mi],  false, br[ni], (short)0, acc_im[mi][ni], false, false);
      }

    __syncthreads();   // all waves done reading stage `buf` before it is refilled
    buf ^= 1;
  }

  // Epilogue: bias + optional elementwise add-in, then store f32 planes.
#pragma unroll
  for (int mi = 0; mi < 2; ++mi)
#pragma unroll
    for (int ni = 0; ni < 2; ++ni) {
      const int ncol = n0 + wn * 32 + ni * 16 + ll;
      float bre = 0.f, bim = 0.f;
      if (b1re) { bre += b1re[ncol]; bim += b1im[ncol]; }
      if (b2re) { bre += b2re[ncol]; bim += b2im[ncol]; }
#pragma unroll
      for (int v = 0; v < 8; ++v) {
        const int mrow = m0 + wm * 32 + mi * 16 + v + 8 * lh;
        const size_t idx = (size_t)mrow * (size_t)N + (size_t)ncol;
        float vre = acc_re[mi][ni][v] + bre;
        float vim = acc_im[mi][ni][v] + bim;
        if (addre) { vre += addre[idx]; vim += addim[idx]; }
        Cre[idx] = vre;
        Cim[idx] = vim;
      }
    }
}

// ---------------------------------------------------------------------------
// fp32 -> bf16 conversion (vectorized x4)
// ---------------------------------------------------------------------------
__global__ __launch_bounds__(256)
void cgru_f32_to_bf16(const float* __restrict__ in, bf16_t* __restrict__ out, int n) {
  int idx = (blockIdx.x * blockDim.x + threadIdx.x) * 4;
  if (idx + 3 < n) {
    const float4 v = *(const float4*)&in[idx];
    out[idx + 0] = (bf16_t)v.x;
    out[idx + 1] = (bf16_t)v.y;
    out[idx + 2] = (bf16_t)v.z;
    out[idx + 3] = (bf16_t)v.w;
  } else {
    for (; idx < n; ++idx) out[idx] = (bf16_t)in[idx];
  }
}

// ---------------------------------------------------------------------------
// z = sigmoid(accz) (written in place, f32); r = sigmoid(accr);
// rh = r * h (complex), emitted as bf16 planes for the final GEMM.
// ---------------------------------------------------------------------------
__global__ __launch_bounds__(256)
void cgru_ew_zr(const float* __restrict__ hre, const float* __restrict__ him,
                float* __restrict__ zre, float* __restrict__ zim,
                const float* __restrict__ rre_pre, const float* __restrict__ rim_pre,
                bf16_t* __restrict__ rhre, bf16_t* __restrict__ rhim, int n) {
  const int i = blockIdx.x * blockDim.x + threadIdx.x;
  if (i >= n) return;
  const float zr = sigmoidf_dev(zre[i]);
  const float zi = sigmoidf_dev(zim[i]);
  zre[i] = zr;
  zim[i] = zi;
  const float rr = sigmoidf_dev(rre_pre[i]);
  const float ri = sigmoidf_dev(rim_pre[i]);
  const float hr = hre[i], hi = him[i];
  rhre[i] = (bf16_t)(rr * hr - ri * hi);
  rhim[i] = (bf16_t)(rr * hi + ri * hr);
}

// ---------------------------------------------------------------------------
// h_tilde = polar_tanh(p); h_new = (1-z)*h + z*h_tilde; out interleaved [B,H,2]
// ---------------------------------------------------------------------------
__global__ __launch_bounds__(256)
void cgru_ew_final(const float* __restrict__ hre, const float* __restrict__ him,
                   const float* __restrict__ zre, const float* __restrict__ zim,
                   const float* __restrict__ pre, const float* __restrict__ pim,
                   float* __restrict__ out, int n) {
  const int i = blockIdx.x * blockDim.x + threadIdx.x;
  if (i >= n) return;
  const float pr = pre[i], pi = pim[i];
  const float mag = sqrtf(pr * pr + pi * pi);
  const float safe = (mag == 0.f) ? 1.f : mag;
  const float s = tanhf(mag) / safe;
  const float htr = s * pr, hti = s * pi;
  const float zr = zre[i], zi = zim[i];
  const float hr = hre[i], hi = him[i];
  const float t1r = (1.f - zr) * hr + zi * hi;
  const float t1i = (1.f - zr) * hi - zi * hr;
  const float t2r = zr * htr - zi * hti;
  const float t2i = zr * hti + zi * htr;
  out[2 * (size_t)i + 0] = t1r + t2r;
  out[2 * (size_t)i + 1] = t1i + t2i;
}

// ---------------------------------------------------------------------------
extern "C" void kernel_launch(void* const* d_in, const int* in_sizes, int n_in,
                              void* d_out, int out_size, void* d_ws, size_t ws_size,
                              hipStream_t stream) {
  const float* x_re  = (const float*)d_in[0];
  const float* x_im  = (const float*)d_in[1];
  const float* h_re  = (const float*)d_in[2];
  const float* h_im  = (const float*)d_in[3];
  const float* Wx_re = (const float*)d_in[4];
  const float* Wx_im = (const float*)d_in[5];
  const float* Wh_re = (const float*)d_in[6];
  const float* Wh_im = (const float*)d_in[7];
  const float* bx_re = (const float*)d_in[8];
  const float* bx_im = (const float*)d_in[9];
  const float* bh_re = (const float*)d_in[10];
  const float* bh_im = (const float*)d_in[11];
  float* out = (float*)d_out;

  const int D = 512, H = 512;
  const int B = in_sizes[0] / D;            // 16384
  const size_t BD   = (size_t)B * D;
  const size_t BH   = (size_t)B * H;
  const size_t WXsz = (size_t)3 * H * D;
  const size_t WHsz = (size_t)3 * H * H;

  uint8_t* wsp = (uint8_t*)d_ws;
  auto carve = [&](size_t bytes) -> void* {
    void* p = (void*)wsp;
    wsp += (bytes + 255) & ~(size_t)255;
    return p;
  };

  bf16_t* xb_re  = (bf16_t*)carve(BD * 2);
  bf16_t* xb_im  = (bf16_t*)carve(BD * 2);
  bf16_t* hb_re  = (bf16_t*)carve(BH * 2);
  bf16_t* hb_im  = (bf16_t*)carve(BH * 2);
  bf16_t* Wxb_re = (bf16_t*)carve(WXsz * 2);
  bf16_t* Wxb_im = (bf16_t*)carve(WXsz * 2);
  bf16_t* Whb_re = (bf16_t*)carve(WHsz * 2);
  bf16_t* Whb_im = (bf16_t*)carve(WHsz * 2);
  float*  accz_re = (float*)carve(BH * 4);
  float*  accz_im = (float*)carve(BH * 4);
  float*  accr_re = (float*)carve(BH * 4);
  float*  accr_im = (float*)carve(BH * 4);
  float*  acch_re = (float*)carve(BH * 4);
  float*  acch_im = (float*)carve(BH * 4);
  bf16_t* rh_re   = (bf16_t*)carve(BH * 2);
  bf16_t* rh_im   = (bf16_t*)carve(BH * 2);

  // --- 1. fp32 -> bf16 conversions -------------------------------------
  auto conv = [&](const float* src, bf16_t* dst, size_t n) {
    const int nb = (int)((n / 4 + 255) / 256);
    cgru_f32_to_bf16<<<nb, 256, 0, stream>>>(src, dst, (int)n);
  };
  conv(x_re, xb_re, BD);   conv(x_im, xb_im, BD);
  conv(h_re, hb_re, BH);   conv(h_im, hb_im, BH);
  conv(Wx_re, Wxb_re, WXsz); conv(Wx_im, Wxb_im, WXsz);
  conv(Wh_re, Whb_re, WHsz); conv(Wh_im, Whb_im, WHsz);

  const dim3 gg(B / TILE_M, H / TILE_N);
  const size_t HD = (size_t)H * D, HH = (size_t)H * H;

  // --- 2. gate z: x@Wx0^T + h@Wh0^T + bx0 + bh0 ------------------------
  cgru_gemm_cplx<<<gg, 256, 0, stream>>>(
      xb_re, xb_im, Wxb_re + 0 * HD, Wxb_im + 0 * HD, D,
      hb_re, hb_im, Whb_re + 0 * HH, Whb_im + 0 * HH, H,
      bx_re + 0 * H, bx_im + 0 * H, bh_re + 0 * H, bh_im + 0 * H,
      nullptr, nullptr, accz_re, accz_im, H);

  // --- 3. gate r -------------------------------------------------------
  cgru_gemm_cplx<<<gg, 256, 0, stream>>>(
      xb_re, xb_im, Wxb_re + 1 * HD, Wxb_im + 1 * HD, D,
      hb_re, hb_im, Whb_re + 1 * HH, Whb_im + 1 * HH, H,
      bx_re + 1 * H, bx_im + 1 * H, bh_re + 1 * H, bh_im + 1 * H,
      nullptr, nullptr, accr_re, accr_im, H);

  // --- 4. px2 = x@Wx2^T + bx2 ------------------------------------------
  cgru_gemm_cplx<<<gg, 256, 0, stream>>>(
      xb_re, xb_im, Wxb_re + 2 * HD, Wxb_im + 2 * HD, D,
      nullptr, nullptr, nullptr, nullptr, 0,
      bx_re + 2 * H, bx_im + 2 * H, nullptr, nullptr,
      nullptr, nullptr, acch_re, acch_im, H);

  // --- 5. activations for z,r + rh = r*h (bf16) ------------------------
  {
    const int nb = (int)((BH + 255) / 256);
    cgru_ew_zr<<<nb, 256, 0, stream>>>(h_re, h_im, accz_re, accz_im,
                                       accr_re, accr_im, rh_re, rh_im, (int)BH);
  }

  // --- 6. acch = px2 + (r*h)@Wh2^T + bh2 -------------------------------
  cgru_gemm_cplx<<<gg, 256, 0, stream>>>(
      rh_re, rh_im, Whb_re + 2 * HH, Whb_im + 2 * HH, H,
      nullptr, nullptr, nullptr, nullptr, 0,
      bh_re + 2 * H, bh_im + 2 * H, nullptr, nullptr,
      acch_re, acch_im, acch_re, acch_im, H);

  // --- 7. polar tanh + gate combine + interleaved store ----------------
  {
    const int nb = (int)((BH + 255) / 256);
    cgru_ew_final<<<nb, 256, 0, stream>>>(h_re, h_im, accz_re, accz_im,
                                          acch_re, acch_im, out, (int)BH);
  }
  (void)n_in; (void)out_size; (void)ws_size;
}
